// CodeGenAttention_7000796692521
// MI455X (gfx1250) — compile-verified, matrix operand order checked by reading, not run on hardware
//
#include <hip/hip_runtime.h>
#include <hip/hip_bf16.h>

#define NH    16
#define HD    256
#define SEQ   2048
#define BATCH 2
#define HID   4096
#define M_TOK (BATCH * SEQ)   // 4096
#define N_QKV (3 * HID)       // 12288

typedef unsigned short u16;
typedef __attribute__((ext_vector_type(16))) __bf16       v16bf;
typedef __attribute__((ext_vector_type(8)))  float        v8f;
typedef __attribute__((ext_vector_type(8)))  unsigned int v8u;
typedef __attribute__((ext_vector_type(4)))  int          i32x4;

union Frag { v8u u; v16bf b; };

// ---------------------------------------------------------------------------
// Async global->LDS copy (CDNA5 GLOBAL_LOAD_ASYNC_TO_LDS_B128, ASYNCcnt),
// with a synchronous fallback if the toolchain lacks the builtin.
// Builtin signature (from hipcc diagnostic): (i32x4 AS1*, i32x4 AS3*, Ii, Ii),
// first param non-const.
// ---------------------------------------------------------------------------
#if __has_builtin(__builtin_amdgcn_global_load_async_to_lds_b128) && \
    __has_builtin(__builtin_amdgcn_s_wait_asynccnt)
#define HAVE_ASYNC_LDS 1
typedef __attribute__((address_space(1))) i32x4* as1_i32x4_p;
typedef __attribute__((address_space(3))) i32x4* as3_i32x4_p;
#define ASYNC_WAIT(n) __builtin_amdgcn_s_wait_asynccnt(n)
__device__ __forceinline__ void cp_async16(void* lds, const void* gmem) {
  __builtin_amdgcn_global_load_async_to_lds_b128(
      (as1_i32x4_p)gmem, (as3_i32x4_p)lds, 0, 0);
}
#else
#define ASYNC_WAIT(n)
__device__ __forceinline__ void cp_async16(void* lds, const void* gmem) {
  *(i32x4*)lds = *(const i32x4*)gmem;
}
#endif

__device__ __forceinline__ u16 f2bf(float f) {
  unsigned u = __builtin_bit_cast(unsigned, f);
  u += 0x7FFFu + ((u >> 16) & 1u);   // round-to-nearest-even
  return (u16)(u >> 16);
}
__device__ __forceinline__ float bf2f(u16 h) {
  return __builtin_bit_cast(float, (unsigned)h << 16);
}

// Load one 16x32 bf16 WMMA operand fragment from a row-major [row][k] array.
// Per ISA 7.12.2 (16-bit A 16x32): row = lane&15, khalf = (lane>>4)*8,
// VGPR v holds K pair k = (v>=4?16:0) + khalf + (v&3)*2.  Contiguous per lane
// -> compiler emits 2x ds_load_b128.  B operand mirrors this on [n][k].
__device__ __forceinline__ v16bf load_frag(const u16* base, int stride) {
  const int lane  = threadIdx.x & 31;
  const int row   = lane & 15;
  const int khalf = (lane >> 4) << 3;
  Frag f;
#pragma unroll
  for (int v = 0; v < 8; ++v) {
    int k = ((v & 4) << 2) + khalf + ((v & 3) << 1);
    f.u[v] = *(const unsigned int*)(base + row * stride + k);
  }
  return f.b;
}

// ---------------------------------------------------------------------------
// fp32 -> bf16 elementwise convert (vectorized x4)
// ---------------------------------------------------------------------------
__global__ __launch_bounds__(256) void f32_to_bf16_kernel(
    const float* __restrict__ in, u16* __restrict__ out, int n) {
  int i = (blockIdx.x * 256 + threadIdx.x) * 4;
  if (i + 3 < n) {
    float4 v = *(const float4*)(in + i);
    out[i + 0] = f2bf(v.x);
    out[i + 1] = f2bf(v.y);
    out[i + 2] = f2bf(v.z);
    out[i + 3] = f2bf(v.w);
  }
}

// ---------------------------------------------------------------------------
// W [K][N] fp32 -> WT [N][K] bf16 (tiled via LDS, conflict-free)
// ---------------------------------------------------------------------------
__global__ __launch_bounds__(256) void transpose_f32_to_bf16_kernel(
    const float* __restrict__ W, u16* __restrict__ WT, int K, int N) {
  __shared__ float tile[32][33];
  int n0 = blockIdx.x * 32, k0 = blockIdx.y * 32;
  int tx = threadIdx.x & 31, ty = threadIdx.x >> 5;  // ty: 0..7
#pragma unroll
  for (int i = ty; i < 32; i += 8)
    tile[i][tx] = W[(size_t)(k0 + i) * N + n0 + tx];
  __syncthreads();
#pragma unroll
  for (int i = ty; i < 32; i += 8)
    WT[(size_t)(n0 + i) * K + k0 + tx] = f2bf(tile[tx][i]);
}

// ---------------------------------------------------------------------------
// bf16 WMMA GEMM: C[M][N] = A[M][K] * Bt[N][K]^T, fp32 accumulate.
// Block tile 128x256, BK=32, double-buffered async LDS staging.
// 8 waves as 2x4; each wave 64x64 = 4x4 fragments (16 WMMA per BK step).
// LDS rows padded to 40 elems (16B) -> conflict-free ds_load_b128 frags.
// ---------------------------------------------------------------------------
#define LDK 40
template <bool OUT_BF16>
__global__ __launch_bounds__(256) void gemm_bf16_wmma(
    const u16* __restrict__ A, const u16* __restrict__ Bt,
    void* __restrict__ C, int M, int N, int K) {
  __shared__ alignas(16) u16 As[2][128 * LDK];  // 2 x 10 KB
  __shared__ alignas(16) u16 Bs[2][256 * LDK];  // 2 x 20 KB
  const int tid  = threadIdx.x;
  const int lane = tid & 31;
  const int wid  = tid >> 5;
  const int m0 = blockIdx.y * 128;
  const int n0 = blockIdx.x * 256;
  const int wm = (wid >> 2) * 64;
  const int wn = (wid & 3) * 64;

  // stage one 128x32 A tile (512 chunks) + 256x32 B tile (1024 chunks):
  // 6 x 16B async loads per thread.
  auto stage = [&](int buf, int k0) {
#pragma unroll
    for (int c = 0; c < 2; ++c) {
      int idx = tid + c * 256;
      int r   = idx >> 2;
      int c8  = (idx & 3) << 3;
      cp_async16(As[buf] + r * LDK + c8, A + (size_t)(m0 + r) * K + k0 + c8);
    }
#pragma unroll
    for (int c = 0; c < 4; ++c) {
      int idx = tid + c * 256;
      int r   = idx >> 2;
      int c8  = (idx & 3) << 3;
      cp_async16(Bs[buf] + r * LDK + c8, Bt + (size_t)(n0 + r) * K + k0 + c8);
    }
  };

  v8f acc[4][4] = {};
  const int KT = K >> 5;
  stage(0, 0);
  for (int kt = 0; kt < KT; ++kt) {
    if (kt + 1 < KT) {
      stage((kt + 1) & 1, (kt + 1) << 5);
      ASYNC_WAIT(6);   // newest 6 may remain in flight; current buffer is done
    } else {
      ASYNC_WAIT(0);
    }
    __syncthreads();
    const u16* Ab = As[kt & 1];
    const u16* Bb = Bs[kt & 1];
    v16bf af[4], bfv[4];
#pragma unroll
    for (int i = 0; i < 4; ++i) af[i]  = load_frag(Ab + (wm + i * 16) * LDK, LDK);
#pragma unroll
    for (int j = 0; j < 4; ++j) bfv[j] = load_frag(Bb + (wn + j * 16) * LDK, LDK);
#pragma unroll
    for (int i = 0; i < 4; ++i)
#pragma unroll
      for (int j = 0; j < 4; ++j)
        acc[i][j] = __builtin_amdgcn_wmma_f32_16x16x32_bf16(
            false, af[i], false, bfv[j], (short)0, acc[i][j], false, false);
    __syncthreads();
  }

  // C layout: lanes 0-15 -> N=lane, M=vgpr; lanes 16-31 -> N=lane-16, M=vgpr+8
  const int nloc = lane & 15;
  const int mh   = (lane >> 4) << 3;
#pragma unroll
  for (int i = 0; i < 4; ++i)
#pragma unroll
    for (int j = 0; j < 4; ++j)
#pragma unroll
      for (int r = 0; r < 8; ++r) {
        size_t row = (size_t)(m0 + wm + i * 16 + mh + r);
        size_t col = (size_t)(n0 + wn + j * 16 + nloc);
        if (OUT_BF16) ((u16*)C)[row * N + col] = f2bf(acc[i][j][r]);
        else          ((float*)C)[row * N + col] = acc[i][j][r];
      }
}

// ---------------------------------------------------------------------------
// Rotary + QKV split.  fused[b,s,12288]: per mp-shard of 3072, order is
// query [0,1024), VALUE [1024,2048), key [2048,3072) (reference split order).
// head h -> shard h/4, sub-head h%4.  Partial rotary on first 64 dims.
// Q pre-scaled by 1/sqrt(256)=1/16.  V written transposed [b,h,d,s].
// ---------------------------------------------------------------------------
__global__ __launch_bounds__(64) void rotary_split_kernel(
    const u16* __restrict__ fused, const int* __restrict__ pos_ids,
    u16* __restrict__ Q, u16* __restrict__ Kd, u16* __restrict__ Vt) {
  int blk     = blockIdx.x;          // [B*S*NH)
  int h       = blk & (NH - 1);
  int tokfull = blk >> 4;            // b*SEQ + s
  int s       = tokfull & (SEQ - 1);
  int b       = tokfull >> 11;
  int d0      = threadIdx.x * 4;
  const float pos = (float)pos_ids[tokfull];
  size_t base = (size_t)tokfull * N_QKV + (size_t)(h >> 2) * 3072 + (size_t)(h & 3) * HD;

  float q[4], k[4], v[4];
#pragma unroll
  for (int e = 0; e < 4; ++e) {
    q[e] = bf2f(fused[base + d0 + e]);
    v[e] = bf2f(fused[base + 1024 + d0 + e]);
    k[e] = bf2f(fused[base + 2048 + d0 + e]);
  }
  if (d0 < 64) {
#pragma unroll
    for (int p = 0; p < 2; ++p) {
      int   j   = (d0 >> 1) + p;
      float inv = __powf(10000.0f, -(float)(2 * j) * (1.0f / 64.0f));
      float si  = pos * inv;
      float sn = __sinf(si), cs = __cosf(si);
      float e0 = q[2 * p], e1 = q[2 * p + 1];
      q[2 * p]     = e0 * cs - e1 * sn;
      q[2 * p + 1] = e1 * cs + e0 * sn;
      e0 = k[2 * p]; e1 = k[2 * p + 1];
      k[2 * p]     = e0 * cs - e1 * sn;
      k[2 * p + 1] = e1 * cs + e0 * sn;
    }
  }
  size_t qk = ((size_t)(b * NH + h) * SEQ + s) * HD + d0;
#pragma unroll
  for (int e = 0; e < 4; ++e) {
    Q[qk + e]  = f2bf(q[e] * 0.0625f);
    Kd[qk + e] = f2bf(k[e]);
    Vt[((size_t)(b * NH + h) * HD + d0 + e) * SEQ + s] = f2bf(v[e]);
  }
}

// ---------------------------------------------------------------------------
// Causal flash attention.  Grid: B*NH*(SEQ/128).  8 waves/block; wave w owns
// query rows [q0+16w, q0+16w+16).  32-key K/V tiles double-buffered in LDS
// via async copies; padded strides for conflict-free fragment loads.
// ---------------------------------------------------------------------------
#define LKS (HD + 8)   // 264
#define LVS (32 + 8)   // 40
__global__ __launch_bounds__(256) void flash_attn_kernel(
    const u16* __restrict__ Q, const u16* __restrict__ Kg,
    const u16* __restrict__ Vt, u16* __restrict__ O) {
  __shared__ alignas(16) u16 Ks[2][32 * LKS];   // [key][dim]   ~33 KB
  __shared__ alignas(16) u16 Vs[2][HD * LVS];   // [dim][key]   ~40 KB
  __shared__ alignas(16) u16 Ps[8][16 * LVS];   // per-wave P scratch ~10 KB

  const int nqb  = SEQ / 128;
  const int b    = blockIdx.x / (NH * nqb);
  const int h    = (blockIdx.x / nqb) % NH;
  const int qb   = blockIdx.x % nqb;
  const int tid  = threadIdx.x;
  const int lane = tid & 31;
  const int w    = tid >> 5;
  const int q0   = qb * 128;
  const int r0   = q0 + w * 16;
  const size_t bh = (size_t)(b * NH + h);

  auto stage_kv = [&](int buf, int k0) {
    // K tile: 32x256 bf16 = 1024 chunks; V tile: 256x32 = 1024 chunks.
#pragma unroll
    for (int c = 0; c < 4; ++c) {
      int idx = tid + c * 256;
      int kr  = idx >> 5;
      int c8  = (idx & 31) << 3;
      cp_async16(Ks[buf] + kr * LKS + c8,
                 Kg + (bh * SEQ + k0 + kr) * HD + c8);
    }
#pragma unroll
    for (int c = 0; c < 4; ++c) {
      int idx = tid + c * 256;
      int dr  = idx >> 2;
      int s8  = (idx & 3) << 3;
      cp_async16(Vs[buf] + dr * LVS + s8,
                 Vt + (bh * HD + dr) * SEQ + k0 + s8);
    }
  };

  // persistent Q fragments: 16 rows x 256 dims = 8 chunks of K=32
  v16bf qf[8];
#pragma unroll
  for (int c = 0; c < 8; ++c)
    qf[c] = load_frag(Q + (bh * SEQ + r0) * HD + c * 32, HD);

  v8f acc[16] = {};
  float mi[8], li[8];
#pragma unroll
  for (int r = 0; r < 8; ++r) { mi[r] = -3.0e30f; li[r] = 0.0f; }

  const int nloc = lane & 15;
  const int mh   = (lane >> 4) << 3;

  const int nkb = (q0 + 128) / 32;
  stage_kv(0, 0);
  for (int kb = 0; kb < nkb; ++kb) {
    const int k0 = kb * 32;
    if (kb + 1 < nkb) {
      stage_kv((kb + 1) & 1, (kb + 1) * 32);
      ASYNC_WAIT(8);   // current buffer complete; next 8 may be in flight
    } else {
      ASYNC_WAIT(0);
    }
    __syncthreads();

    if (k0 <= r0 + 15) {  // wave-uniform causal block skip
      const u16* Kb = Ks[kb & 1];
      const u16* Vb = Vs[kb & 1];
      // S = Q * K^T : two 16x16 score fragments, K-depth 256
      v8f sfr[2] = {};
#pragma unroll
      for (int f = 0; f < 2; ++f)
#pragma unroll
        for (int c = 0; c < 8; ++c) {
          v16bf kfrag = load_frag(Kb + (f * 16) * LKS + c * 32, LKS);
          sfr[f] = __builtin_amdgcn_wmma_f32_16x16x32_bf16(
              false, qf[c], false, kfrag, (short)0, sfr[f], false, false);
        }
      // causal mask + online softmax (rows live across the 16-lane half-wave)
      float p[2][8];
#pragma unroll
      for (int r = 0; r < 8; ++r) {
        int row = r0 + mh + r;
        float t = -3.0e30f;
#pragma unroll
        for (int f = 0; f < 2; ++f) {
          int key  = k0 + f * 16 + nloc;
          float sv = (key <= row) ? sfr[f][r] : -3.0e30f;
          p[f][r] = sv;
          t = fmaxf(t, sv);
        }
        t = fmaxf(t, __shfl_xor(t, 1));
        t = fmaxf(t, __shfl_xor(t, 2));
        t = fmaxf(t, __shfl_xor(t, 4));
        t = fmaxf(t, __shfl_xor(t, 8));
        float newm = fmaxf(mi[r], t);
        float rs = 0.0f;
#pragma unroll
        for (int f = 0; f < 2; ++f) {
          float e = __expf(p[f][r] - newm);
          p[f][r] = e;
          rs += e;
        }
        rs += __shfl_xor(rs, 1);
        rs += __shfl_xor(rs, 2);
        rs += __shfl_xor(rs, 4);
        rs += __shfl_xor(rs, 8);
        float alpha = __expf(mi[r] - newm);
        li[r] = li[r] * alpha + rs;
        mi[r] = newm;
#pragma unroll
        for (int j = 0; j < 16; ++j) acc[j][r] *= alpha;
      }
      // re-layout P (C-frag -> A-frag) through per-wave LDS scratch
#pragma unroll
      for (int f = 0; f < 2; ++f)
#pragma unroll
        for (int r = 0; r < 8; ++r)
          Ps[w][(mh + r) * LVS + f * 16 + nloc] = f2bf(p[f][r]);
      __builtin_amdgcn_wave_barrier();  // keep ds stores before ds loads
      v16bf pf = load_frag(&Ps[w][0], LVS);
      // acc += P (16x32) * V (32x256)
#pragma unroll
      for (int j = 0; j < 16; ++j) {
        v16bf vfrag = load_frag(Vb + (j * 16) * LVS, LVS);
        acc[j] = __builtin_amdgcn_wmma_f32_16x16x32_bf16(
            false, pf, false, vfrag, (short)0, acc[j], false, false);
      }
    }
    __syncthreads();
  }

  // O[b*SEQ+row][h*256 + col] (bf16, A-matrix for out-proj GEMM)
#pragma unroll
  for (int j = 0; j < 16; ++j)
#pragma unroll
    for (int r = 0; r < 8; ++r) {
      size_t row = (size_t)(r0 + mh + r);
      size_t col = (size_t)(h * HD + j * 16 + nloc);
      O[((size_t)b * SEQ + row) * HID + col] = f2bf(acc[j][r] / li[r]);
    }
}

// ---------------------------------------------------------------------------
extern "C" void kernel_launch(void* const* d_in, const int* in_sizes, int n_in,
                              void* d_out, int out_size, void* d_ws, size_t ws_size,
                              hipStream_t stream) {
  (void)in_sizes; (void)n_in; (void)out_size; (void)ws_size;
  const float* hs   = (const float*)d_in[0];
  const int*   pos  = (const int*)d_in[1];
  const float* wqkv = (const float*)d_in[3];  // d_in[2] = attention_mask (all true)
  const float* wout = (const float*)d_in[4];

  char* ws = (char*)d_ws;
  size_t off = 0;
  auto alloc = [&](size_t bytes) -> void* {
    void* p = ws + off;
    off = (off + bytes + 255) & ~(size_t)255;
    return p;
  };
  u16* Xbf   = (u16*)alloc((size_t)M_TOK * HID * 2);           // 32 MB
  u16* WqkvT = (u16*)alloc((size_t)N_QKV * HID * 2);           // 96 MB
  u16* WoutT = (u16*)alloc((size_t)HID * HID * 2);             // 32 MB
  u16* Fused = (u16*)alloc((size_t)M_TOK * N_QKV * 2);         // 96 MB
  u16* Qb    = (u16*)alloc((size_t)BATCH * NH * SEQ * HD * 2); // 32 MB
  u16* Kb    = (u16*)alloc((size_t)BATCH * NH * SEQ * HD * 2);
  u16* Vtb   = (u16*)alloc((size_t)BATCH * NH * SEQ * HD * 2);
  u16* Ob    = (u16*)alloc((size_t)M_TOK * HID * 2);

  f32_to_bf16_kernel<<<(M_TOK * HID) / 1024, 256, 0, stream>>>(hs, Xbf, M_TOK * HID);
  transpose_f32_to_bf16_kernel<<<dim3(N_QKV / 32, HID / 32), 256, 0, stream>>>(
      wqkv, WqkvT, HID, N_QKV);
  transpose_f32_to_bf16_kernel<<<dim3(HID / 32, HID / 32), 256, 0, stream>>>(
      wout, WoutT, HID, HID);

  gemm_bf16_wmma<true><<<dim3(N_QKV / 256, M_TOK / 128), 256, 0, stream>>>(
      Xbf, WqkvT, Fused, M_TOK, N_QKV, HID);

  rotary_split_kernel<<<BATCH * SEQ * NH, 64, 0, stream>>>(Fused, pos, Qb, Kb, Vtb);

  flash_attn_kernel<<<BATCH * NH * (SEQ / 128), 256, 0, stream>>>(Qb, Kb, Vtb, Ob);

  gemm_bf16_wmma<false><<<dim3(HID / 256, M_TOK / 128), 256, 0, stream>>>(
      Ob, WoutT, d_out, M_TOK, HID, HID);
}
